// Attention_15539191677265
// MI455X (gfx1250) — matrix operand order repaired and checked by reading, not compiled, not run on hardware
//
#include <hip/hip_runtime.h>
#include <hip/hip_bf16.h>

typedef __attribute__((ext_vector_type(16))) _Float16 v16h;
typedef __attribute__((ext_vector_type(8)))  float    v8f;

#define HEADS 8
#define DIMH  64
#define SEQ   2048
#define CIN   256
#define HID   512
#define BATCH 8
#define SCALE 0.125f  // 64^-0.5

__device__ __forceinline__ v8f wmma16(v16h a, v16h b, v8f c) {
  return __builtin_amdgcn_wmma_f32_16x16x32_f16(false, a, false, b, (short)0, c, false, false);
}

// A operand, 16x32 f16, memory row-major [16 rows, ld], rows = M, cols = K.
// ISA 7.12.2: lanes 0-15 -> M=lane, K-pair offset 0; lanes 16-31 -> M=lane-16, offset 8.
__device__ __forceinline__ v16h load_a(const _Float16* __restrict__ base, int ld, int lane) {
  v16h a;
  const int m  = lane & 15;
  const int kb = (lane < 16) ? 0 : 8;
  const _Float16* p = base + (size_t)m * ld;
#pragma unroll
  for (int r = 0; r < 4; ++r) { a[2*r]   = p[kb + 2*r];       a[2*r+1]   = p[kb + 2*r + 1]; }
#pragma unroll
  for (int r = 0; r < 4; ++r) { a[8+2*r] = p[16 + kb + 2*r];  a[8+2*r+1] = p[16 + kb + 2*r + 1]; }
  return a;
}

// B operand, 32x16 f16. Memory holds value(k,n) at base[n*ld + k] (pairs contiguous in k).
// lanes 0-15 -> N=lane, K=0..7,16..23; lanes 16-31 -> N=lane-16, K=8..15,24..31.
__device__ __forceinline__ v16h load_b(const _Float16* __restrict__ base, int ld, int lane) {
  v16h b;
  const int n  = lane & 15;
  const int kb = (lane < 16) ? 0 : 8;
  const _Float16* p = base + (size_t)n * ld + kb;
#pragma unroll
  for (int r = 0; r < 8; ++r) { b[2*r] = p[2*r + (r >> 2) * 8]; b[2*r+1] = p[2*r+1 + (r >> 2) * 8]; }
  return b;
}

__device__ __forceinline__ float rowmax16(float v) {
#pragma unroll
  for (int off = 1; off < 16; off <<= 1) v = fmaxf(v, __shfl_xor(v, off, 32));
  return v;
}
__device__ __forceinline__ float rowsum16(float v) {
#pragma unroll
  for (int off = 1; off < 16; off <<= 1) v += __shfl_xor(v, off, 32);
  return v;
}

// ---------------- conversion kernels ----------------
__global__ void cvt_x_kernel(const float* __restrict__ x, _Float16* __restrict__ xt) {
  size_t i = (size_t)blockIdx.x * blockDim.x + threadIdx.x;
  const size_t N = (size_t)BATCH * CIN * SEQ;
  if (i >= N) return;
  int w = (int)(i % SEQ);
  int c = (int)((i / SEQ) % CIN);
  int b = (int)(i / ((size_t)SEQ * CIN));
  xt[((size_t)b * SEQ + w) * CIN + c] = (_Float16)x[i];
}

__global__ void cvt_f32_f16_kernel(const float* __restrict__ src, _Float16* __restrict__ dst, int n) {
  int i = blockIdx.x * blockDim.x + threadIdx.x;
  if (i < n) dst[i] = (_Float16)src[i];
}

// ---------------- QKV projection ----------------
// One wave computes a 32x32 tile of qkv[o,x] = sum_c w_qkv[o,c] * x[b,c,x].
// Q -> [b,h,w,64] (scaled), K -> [b,h,w,64], V -> [b,h,64,w], all f16.
__global__ void qkv_gemm_kernel(const _Float16* __restrict__ Wh,   // [1536,256]
                                const _Float16* __restrict__ Xt,   // [b,2048,256]
                                _Float16* __restrict__ Q,
                                _Float16* __restrict__ Kb,
                                _Float16* __restrict__ Vb) {
  const int lane = threadIdx.x & 31;
  const int wave = (blockIdx.x * blockDim.x + threadIdx.x) >> 5;
  const int ntile = wave & 63;            // 2048/32
  const int t     = wave >> 6;
  const int mtile = t % 48;               // 1536/32
  const int b     = t / 48;

  const _Float16* A = Wh + (size_t)mtile * 32 * CIN;
  const _Float16* B = Xt + ((size_t)b * SEQ + (size_t)ntile * 32) * CIN;

  v8f acc[2][2] = {};
#pragma unroll 2
  for (int k = 0; k < CIN; k += 32) {
    __builtin_prefetch(B + k + 64, 0, 1);
    v16h a0 = load_a(A + k,                     CIN, lane);
    v16h a1 = load_a(A + (size_t)16 * CIN + k,  CIN, lane);
    v16h b0 = load_b(B + k,                     CIN, lane);
    v16h b1 = load_b(B + (size_t)16 * CIN + k,  CIN, lane);
    acc[0][0] = wmma16(a0, b0, acc[0][0]);
    acc[0][1] = wmma16(a0, b1, acc[0][1]);
    acc[1][0] = wmma16(a1, b0, acc[1][0]);
    acc[1][1] = wmma16(a1, b1, acc[1][1]);
  }

  const int col  = lane & 15;
  const int rofs = (lane < 16) ? 0 : 8;
#pragma unroll
  for (int mt2 = 0; mt2 < 2; ++mt2) {
#pragma unroll
    for (int nt2 = 0; nt2 < 2; ++nt2) {
      const int x = ntile * 32 + nt2 * 16 + col;
#pragma unroll
      for (int g = 0; g < 8; ++g) {
        int o = mtile * 32 + mt2 * 16 + rofs + g;
        float v = acc[mt2][nt2][g];
        if (o < HID) {                       // Q (scaled), [b,h,w,64]
          int h = o >> 6, dh = o & 63;
          Q[(((size_t)(b * HEADS + h) * SEQ) + x) * DIMH + dh] = (_Float16)(v * SCALE);
        } else if (o < 2 * HID) {            // K, [b,h,w,64]
          int oo = o - HID; int h = oo >> 6, dh = oo & 63;
          Kb[(((size_t)(b * HEADS + h) * SEQ) + x) * DIMH + dh] = (_Float16)v;
        } else {                             // V, [b,h,64,w]
          int oo = o - 2 * HID; int h = oo >> 6, dh = oo & 63;
          Vb[(((size_t)(b * HEADS + h) * DIMH) + dh) * SEQ + x] = (_Float16)v;
        }
      }
    }
  }
}

// ---------------- flash attention ----------------
// One wave per (b,h,32-query tile); online softmax over 2048 keys, 32-key steps.
__global__ void flash_attn_kernel(const _Float16* __restrict__ Q,
                                  const _Float16* __restrict__ Kb,
                                  const _Float16* __restrict__ Vb,
                                  _Float16* __restrict__ Ob) {   // [b, w, 512]
  __shared__ _Float16 pstage[8][32 * 32];   // per-wave P staging (C/D -> A layout)
  const int lane  = threadIdx.x & 31;
  const int wslot = threadIdx.x >> 5;
  const int wave  = (blockIdx.x * blockDim.x + threadIdx.x) >> 5;
  const int qtile = wave & 63;              // 2048/32
  const int bh    = wave >> 6;
  const int h     = bh & (HEADS - 1);
  const int b     = bh >> 3;

  const _Float16* Qp = Q  + (((size_t)(b * HEADS + h) * SEQ) + (size_t)qtile * 32) * DIMH;
  const _Float16* Kp = Kb + ((size_t)(b * HEADS + h) * SEQ) * DIMH;
  const _Float16* Vp = Vb + ((size_t)(b * HEADS + h) * DIMH) * SEQ;

  v16h aq[2][2];                            // [query-row tile][d half]
#pragma unroll
  for (int r = 0; r < 2; ++r) {
    aq[r][0] = load_a(Qp + (size_t)r * 16 * DIMH,      DIMH, lane);
    aq[r][1] = load_a(Qp + (size_t)r * 16 * DIMH + 32, DIMH, lane);
  }

  v8f O[2][4] = {};
  float mrun[2][8], lrun[2][8];
#pragma unroll
  for (int r = 0; r < 2; ++r)
#pragma unroll
    for (int g = 0; g < 8; ++g) { mrun[r][g] = -3.0e38f; lrun[r][g] = 0.0f; }

  _Float16* pt = pstage[wslot];
  const int col  = lane & 15;
  const int rofs = (lane < 16) ? 0 : 8;

  for (int j = 0; j < SEQ; j += 32) {
    __builtin_prefetch(Kp + (size_t)(j + 32) * DIMH, 0, 1);
    __builtin_prefetch(Vp + j + 32, 0, 1);

    // S = Q*K^T : key-fragments shared across both query-row tiles
    v16h bk00 = load_b(Kp + (size_t)j * DIMH,             DIMH, lane);
    v16h bk01 = load_b(Kp + (size_t)j * DIMH + 32,        DIMH, lane);
    v16h bk10 = load_b(Kp + (size_t)(j + 16) * DIMH,      DIMH, lane);
    v16h bk11 = load_b(Kp + (size_t)(j + 16) * DIMH + 32, DIMH, lane);

    v8f S[2][2] = {};
#pragma unroll
    for (int r = 0; r < 2; ++r) {
      S[r][0] = wmma16(aq[r][0], bk00, S[r][0]);
      S[r][0] = wmma16(aq[r][1], bk01, S[r][0]);
      S[r][1] = wmma16(aq[r][0], bk10, S[r][1]);
      S[r][1] = wmma16(aq[r][1], bk11, S[r][1]);
    }

    // online softmax: fused reductions (one max + one sum per row)
#pragma unroll
    for (int r = 0; r < 2; ++r) {
#pragma unroll
      for (int g = 0; g < 8; ++g) {
        float mt = rowmax16(fmaxf(S[r][0][g], S[r][1][g]));
        float mn = fmaxf(mrun[r][g], mt);
        float alpha = __expf(mrun[r][g] - mn);
        float p0 = __expf(S[r][0][g] - mn);
        float p1 = __expf(S[r][1][g] - mn);
        lrun[r][g] = lrun[r][g] * alpha + rowsum16(p0 + p1);
        mrun[r][g] = mn;
        S[r][0][g] = p0; S[r][1][g] = p1;
#pragma unroll
        for (int t = 0; t < 4; ++t) O[r][t][g] *= alpha;
      }
    }

    // P (32x32) : C/D layout -> LDS row-major -> A layout
#pragma unroll
    for (int r = 0; r < 2; ++r)
#pragma unroll
      for (int g = 0; g < 8; ++g) {
        pt[(r * 16 + rofs + g) * 32 + col]      = (_Float16)S[r][0][g];
        pt[(r * 16 + rofs + g) * 32 + 16 + col] = (_Float16)S[r][1][g];
      }
    v16h ap0 = load_a(pt,            32, lane);
    v16h ap1 = load_a(pt + 16 * 32,  32, lane);

    // O += P * V : V as [64, 2048]; each V fragment reused by both row tiles
#pragma unroll
    for (int t = 0; t < 4; ++t) {
      v16h bv = load_b(Vp + (size_t)(t * 16) * SEQ + j, SEQ, lane);
      O[0][t] = wmma16(ap0, bv, O[0][t]);
      O[1][t] = wmma16(ap1, bv, O[1][t]);
    }
  }

  // finalize: O /= l ; store Ob[b, x, h*64 + dh] f16
#pragma unroll
  for (int r = 0; r < 2; ++r)
#pragma unroll
    for (int g = 0; g < 8; ++g) {
      float inv = 1.0f / lrun[r][g];
      int x = qtile * 32 + r * 16 + rofs + g;
      size_t rb = ((size_t)b * SEQ + x) * HID + (size_t)h * DIMH;
#pragma unroll
      for (int t = 0; t < 4; ++t)
        Ob[rb + t * 16 + col] = (_Float16)(O[r][t][g] * inv);
    }
}

// ---------------- output projection ----------------
// One wave computes a 32x32 tile of out[b,o,x] = sum_c w_out[o,c]*Ob[b,x,c] + bias[o]
__global__ void out_gemm_kernel(const _Float16* __restrict__ Wh,  // [256,512]
                                const _Float16* __restrict__ Ob,  // [b,2048,512]
                                const float* __restrict__ bias,
                                float* __restrict__ out) {
  const int lane = threadIdx.x & 31;
  const int wave = (blockIdx.x * blockDim.x + threadIdx.x) >> 5;
  const int ntile = wave & 63;            // 2048/32
  const int t     = wave >> 6;
  const int mtile = t & 7;                // 256/32
  const int b     = t >> 3;

  const _Float16* A = Wh + (size_t)mtile * 32 * HID;
  const _Float16* B = Ob + ((size_t)b * SEQ + (size_t)ntile * 32) * HID;

  v8f acc[2][2] = {};
#pragma unroll 2
  for (int k = 0; k < HID; k += 32) {
    __builtin_prefetch(B + k + 64, 0, 1);
    v16h a0 = load_a(A + k,                    HID, lane);
    v16h a1 = load_a(A + (size_t)16 * HID + k, HID, lane);
    v16h b0 = load_b(B + k,                    HID, lane);
    v16h b1 = load_b(B + (size_t)16 * HID + k, HID, lane);
    acc[0][0] = wmma16(a0, b0, acc[0][0]);
    acc[0][1] = wmma16(a0, b1, acc[0][1]);
    acc[1][0] = wmma16(a1, b0, acc[1][0]);
    acc[1][1] = wmma16(a1, b1, acc[1][1]);
  }

  const int col  = lane & 15;
  const int rofs = (lane < 16) ? 0 : 8;
#pragma unroll
  for (int mt2 = 0; mt2 < 2; ++mt2) {
#pragma unroll
    for (int nt2 = 0; nt2 < 2; ++nt2) {
      const int x = ntile * 32 + nt2 * 16 + col;
#pragma unroll
      for (int g = 0; g < 8; ++g) {
        int o = mtile * 32 + mt2 * 16 + rofs + g;
        out[((size_t)b * CIN + o) * SEQ + x] = acc[mt2][nt2][g] + bias[o];
      }
    }
  }
}

extern "C" void kernel_launch(void* const* d_in, const int* in_sizes, int n_in,
                              void* d_out, int out_size, void* d_ws, size_t ws_size,
                              hipStream_t stream) {
  const float* x     = (const float*)d_in[0];   // [8,256,2048]
  const float* wqkv  = (const float*)d_in[1];   // [1536,256]
  const float* wout  = (const float*)d_in[2];   // [256,512]
  const float* bout  = (const float*)d_in[3];   // [256]
  float* out = (float*)d_out;

  char* ws = (char*)d_ws;
  size_t off = 0;
  _Float16* xt     = (_Float16*)(ws + off); off += (size_t)BATCH * SEQ * CIN * 2;
  _Float16* wqkv_h = (_Float16*)(ws + off); off += (size_t)3 * HID * CIN * 2;
  _Float16* wout_h = (_Float16*)(ws + off); off += (size_t)CIN * HID * 2;
  _Float16* Qb     = (_Float16*)(ws + off); off += (size_t)BATCH * HEADS * SEQ * DIMH * 2;
  _Float16* Kb     = (_Float16*)(ws + off); off += (size_t)BATCH * HEADS * SEQ * DIMH * 2;
  _Float16* Vb     = (_Float16*)(ws + off); off += (size_t)BATCH * HEADS * SEQ * DIMH * 2;
  _Float16* Ob     = (_Float16*)(ws + off); off += (size_t)BATCH * SEQ * HID * 2;

  // 1. conversions
  {
    size_t n = (size_t)BATCH * CIN * SEQ;
    cvt_x_kernel<<<(unsigned)((n + 255) / 256), 256, 0, stream>>>(x, xt);
    int nq = 3 * HID * CIN;
    cvt_f32_f16_kernel<<<(nq + 255) / 256, 256, 0, stream>>>(wqkv, wqkv_h, nq);
    int no = CIN * HID;
    cvt_f32_f16_kernel<<<(no + 255) / 256, 256, 0, stream>>>(wout, wout_h, no);
  }

  // 2. QKV projection: 32x32 tiles -> waves = 8 * 48 * 64 = 24576 ; 8 waves / block
  qkv_gemm_kernel<<<24576 / 8, 256, 0, stream>>>(wqkv_h, xt, Qb, Kb, Vb);

  // 3. flash attention: 32-query tiles -> waves = 8 * 8 * 64 = 4096
  flash_attn_kernel<<<4096 / 8, 256, 0, stream>>>(Qb, Kb, Vb, Ob);

  // 4. output projection: 32x32 tiles -> waves = 8 * 8 * 64 = 4096
  out_gemm_kernel<<<4096 / 8, 256, 0, stream>>>(wout_h, Ob, bout, out);
}
